// MoELayer_19413252178271
// MI455X (gfx1250) — matrix operand order repaired and checked by reading, not compiled, not run on hardware
//
#include <hip/hip_runtime.h>
#include <hip/hip_bf16.h>
#include <cstdint>

// ---------------------------------------------------------------------------
// MoE + FlyLoRA (top-2 of 8 experts), gather-based routing, f16 WMMA GEMMs.
// H=1024, I=2048, E=8, R=16, T=B*S=2048, TOP_K=2
// BM=64 per block (4 M-subtiles/wave); LDS pitch 40 halfs (bank-conflict-free
// b128 fragment loads); TDM async tensor load feeds the down-GEMM A operand.
// ---------------------------------------------------------------------------

#define HDIM 1024
#define IDIM 2048
#define NEXP 8
#define RRANK 16
#define NTOK 2048
#define NTOPK 2
#define SCALING_F 2.0f
#define GAMMA_F 0.1f
#define BM 64
#define MSUB 4    // BM / 16
#define LDP 40    // LDS pitch in halfs: 32 data + 8 pad (80 B = 20 banks)

typedef __attribute__((ext_vector_type(16))) _Float16 v16h;
typedef __attribute__((ext_vector_type(8)))  float    v8f;
typedef __attribute__((ext_vector_type(4)))  unsigned int v4u;
typedef __attribute__((ext_vector_type(8)))  int      v8i;
typedef __attribute__((ext_vector_type(4)))  int      v4i;

__device__ __forceinline__ v8f wmma_f16(v16h a, v16h b, v8f c) {
  // D = A(16x32 f16) x B(32x16 f16) + C(16x16 f32)
  return __builtin_amdgcn_wmma_f32_16x16x32_f16(
      /*neg_a=*/false, a, /*neg_b=*/false, b,
      /*c_mod=*/(short)0, c, /*reuse_a=*/false, /*reuse_b=*/false);
}

// A tile in LDS: rows x LDP halfs (32 data + pad), row-major.
// Lane l<16: row=l, K = 0..15; lane l>=16: row=l-16, K = 16..31.
__device__ __forceinline__ v16h load_a_frag(const _Float16* sA, int lane) {
  int row = lane & 15;
  int hi  = lane >> 4;
  const _Float16* p = sA + row * LDP + hi * 16;
  v16h a;
#pragma unroll
  for (int j = 0; j < 8; ++j) a[j] = p[j];
#pragma unroll
  for (int j = 0; j < 8; ++j) a[8 + j] = p[8 + j];
  return a;
}

// B tile in LDS: column-major, 16 cols x 32 k, col pitch LDP halfs.
// Lane l<16: col=l, K=0..15; lane l>=16: col=l-16, K=16..31.
__device__ __forceinline__ v16h load_b_frag(const _Float16* sB, int lane) {
  int col = lane & 15;
  int hi  = lane >> 4;
  const _Float16* p = sB + col * LDP + hi * 16;
  v16h b;
#pragma unroll
  for (int j = 0; j < 16; ++j) b[j] = p[j];
  return b;
}

// ERA(x) = gelu_tanh(x) + gamma * softplus(x)
__device__ __forceinline__ float era_f(float v) {
  float t  = 0.7978845608028654f * (v + 0.044715f * v * v * v);
  float g  = 0.5f * v * (1.0f + tanhf(t));
  float sp = (v > 20.0f) ? v : log1pf(__expf(v));
  return g + GAMMA_F * sp;
}

// ---------------------------------------------------------------------------
// Kernel 1: router. One wave (32 lanes) per token. grid = T/8, block = 256.
// ---------------------------------------------------------------------------
__global__ __launch_bounds__(256) void router_kernel(
    const float* __restrict__ x, const float* __restrict__ rw,
    const float* __restrict__ rb,
    int* counts, int* token_idx, float* token_w,
    int* pr_e, int* pr_slot, float* probs) {
  int wv   = threadIdx.x >> 5;
  int lane = threadIdx.x & 31;
  int t    = blockIdx.x * 8 + wv;
  if (t >= NTOK) return;

  float logit[NEXP];
#pragma unroll
  for (int e = 0; e < NEXP; ++e) {
    float acc = 0.0f;
    for (int h = lane; h < HDIM; h += 32)
      acc += x[(size_t)t * HDIM + h] * rw[(size_t)h * NEXP + e];
#pragma unroll
    for (int off = 16; off > 0; off >>= 1) acc += __shfl_xor(acc, off, 32);
    logit[e] = acc + rb[e];
  }
  if (lane == 0) {
    float mx = logit[0];
#pragma unroll
    for (int e = 1; e < NEXP; ++e) mx = fmaxf(mx, logit[e]);
    float p[NEXP]; float se = 0.0f;
#pragma unroll
    for (int e = 0; e < NEXP; ++e) { p[e] = __expf(logit[e] - mx); se += p[e]; }
    float inv = 1.0f / se;
#pragma unroll
    for (int e = 0; e < NEXP; ++e) { p[e] *= inv; probs[t * NEXP + e] = p[e]; }
    // top-2 (lowest index wins ties, matching jax.lax.top_k)
    int i1 = 0;
#pragma unroll
    for (int e = 1; e < NEXP; ++e) if (p[e] > p[i1]) i1 = e;
    int i2 = (i1 == 0) ? 1 : 0;
#pragma unroll
    for (int e = 0; e < NEXP; ++e) if (e != i1 && p[e] > p[i2]) i2 = e;
    float s = p[i1] + p[i2];
    int s1 = atomicAdd(&counts[i1], 1);
    token_idx[i1 * NTOK + s1] = t;
    token_w[i1 * NTOK + s1]   = p[i1] / s;
    pr_e[t * 2 + 0] = i1; pr_slot[t * 2 + 0] = s1;
    int s2 = atomicAdd(&counts[i2], 1);
    token_idx[i2 * NTOK + s2] = t;
    token_w[i2 * NTOK + s2]   = p[i2] / s;
    pr_e[t * 2 + 1] = i2; pr_slot[t * 2 + 1] = s2;
  }
}

// ---------------------------------------------------------------------------
// Kernel 2: exclusive scan of counts + deterministic aux loss.
// ---------------------------------------------------------------------------
__global__ void finalize_kernel(const int* counts, const float* probs,
                                int* row_off, float* aux_out) {
  __shared__ float ps_sh[NEXP];
  int lane = threadIdx.x;
  if (lane < NEXP) {
    float ps = 0.0f;
    for (int t = 0; t < NTOK; ++t) ps += probs[t * NEXP + lane];
    ps_sh[lane] = ps;
  }
  __syncthreads();
  if (lane == 0) {
    int off = 0; float aux = 0.0f;
    for (int e = 0; e < NEXP; ++e) {
      row_off[e] = off; off += counts[e];
      float f = (float)counts[e] / (float)(NTOK * NTOPK);
      aux += f * (ps_sh[e] / (float)NTOK);
    }
    *aux_out = (float)NEXP * aux;
  }
}

// ---------------------------------------------------------------------------
// Kernel 3: h = ERA(gate_pre) * up_pre  (FlyLoRA base + rank-16 LoRA, fused).
// grid = (IDIM/128, NTOK/BM, NEXP), block = 256 (8 waves; each owns a 16-col
// slice and 4 M-subtiles).  Per wave per K-step: 4 gate-base + 4 up-base
// WMMAs + 1 LoRA down-projection WMMA (waves 0-3 gate, waves 4-7 up); the
// rank-16 LoRA up-projection adds 8 zero-padded WMMAs per wave in the tail.
// ---------------------------------------------------------------------------
__global__ __launch_bounds__(256) void ffn_hmid_kernel(
    const float* __restrict__ x,
    const float* __restrict__ gbw, const float* __restrict__ gbb,
    const float* __restrict__ gdw, const float* __restrict__ gdb,
    const float* __restrict__ guw, const float* __restrict__ gub,
    const float* __restrict__ ubw, const float* __restrict__ ubb,
    const float* __restrict__ udw, const float* __restrict__ udb,
    const float* __restrict__ uuw, const float* __restrict__ uub,
    const int* __restrict__ counts, const int* __restrict__ row_off,
    const int* __restrict__ token_idx, _Float16* __restrict__ hmid) {
  int e   = blockIdx.z;
  int cnt = counts[e];
  int m0  = blockIdx.y * BM;
  if (m0 >= cnt) return;
  int n0   = blockIdx.x * 128;
  int tid  = threadIdx.x;
  int lane = tid & 31;
  int wv   = tid >> 5;

  __shared__ _Float16 sA[BM * LDP];
  __shared__ _Float16 sBg[8][16 * LDP];
  __shared__ _Float16 sBu[8][16 * LDP];
  __shared__ _Float16 sDg[16 * LDP];
  __shared__ _Float16 sDu[16 * LDP];
  __shared__ _Float16 sLg[BM * LDP];
  __shared__ _Float16 sLu[BM * LDP];
  __shared__ int sTok[BM];

  for (int i = tid; i < BM; i += 256) {
    int m = m0 + i;
    sTok[i] = (m < cnt) ? token_idx[e * NTOK + m] : 0;
  }
  __syncthreads();

  const float* Wg = gbw + (size_t)e * HDIM * IDIM;
  const float* Wu = ubw + (size_t)e * HDIM * IDIM;
  const float* Dg = gdw + (size_t)e * HDIM * RRANK;
  const float* Du = udw + (size_t)e * HDIM * RRANK;

  // Wave-uniform LoRA role: waves 0-3 own gate L-subtile (wv), 4-7 own up.
  const _Float16* sDl = (wv < MSUB) ? sDg : sDu;
  const _Float16* sAl = sA + (wv & (MSUB - 1)) * 16 * LDP;

  v8f cg[MSUB]; v8f cu[MSUB]; v8f cl = {};
#pragma unroll
  for (int s = 0; s < MSUB; ++s) { cg[s] = (v8f){}; cu[s] = (v8f){}; }

  for (int k0 = 0; k0 < HDIM; k0 += 32) {
    // Prefetch next weight chunk into cache (global_prefetch_b8).
    if (k0 + 32 < HDIM) {
      const float* pg = &Wg[(size_t)(k0 + 32 + (tid >> 3)) * IDIM + n0 + (tid & 7) * 16];
      const float* pu = &Wu[(size_t)(k0 + 32 + (tid >> 3)) * IDIM + n0 + (tid & 7) * 16];
      __builtin_prefetch(pg, 0, 1);
      __builtin_prefetch(pu, 0, 1);
    }
    // Stage A: BM tokens x 32 K (f32 -> f16), gathered rows.
    for (int idx = tid; idx < BM * 32; idx += 256) {
      int r = idx >> 5, k = idx & 31;
      float v = 0.0f;
      if (m0 + r < cnt) v = x[(size_t)sTok[r] * HDIM + k0 + k];
      sA[r * LDP + k] = (_Float16)v;
    }
    // Stage Bg/Bu: 32 K x 128 N, column-major per 16-col wave chunk.
    for (int idx = tid; idx < 32 * 128; idx += 256) {
      int n = idx & 127, k = idx >> 7;
      int wn = n >> 4, ni = n & 15;
      sBg[wn][ni * LDP + k] = (_Float16)Wg[(size_t)(k0 + k) * IDIM + n0 + n];
      sBu[wn][ni * LDP + k] = (_Float16)Wu[(size_t)(k0 + k) * IDIM + n0 + n];
    }
    // Stage LoRA down projections: 32 K x 16 R, column-major.
    for (int idx = tid; idx < 32 * 16; idx += 256) {
      int j = idx & 15, k = idx >> 4;
      sDg[j * LDP + k] = (_Float16)Dg[(size_t)(k0 + k) * RRANK + j];
      sDu[j * LDP + k] = (_Float16)Du[(size_t)(k0 + k) * RRANK + j];
    }
    __syncthreads();

    v16h bgf = load_b_frag(sBg[wv], lane);
    v16h buf = load_b_frag(sBu[wv], lane);
#pragma unroll
    for (int s = 0; s < MSUB; ++s) {
      v16h a = load_a_frag(sA + s * 16 * LDP, lane);
      cg[s] = wmma_f16(a, bgf, cg[s]);
      cu[s] = wmma_f16(a, buf, cu[s]);
    }
    // LoRA down projection: reload this wave's A subtile from LDS (address
    // select, not register select) to avoid cndmask chains.
    cl = wmma_f16(load_a_frag(sAl, lane), load_b_frag(sDl, lane), cl);
    __syncthreads();
  }

  // Spill L tiles (+down bias) into A-fragment layout; each wave writes its
  // own 16x16 subtile plus its K=16..31 zero padding.
  {
    _Float16* sL = (wv < MSUB) ? sLg : sLu;
    const float* dbp = (wv < MSUB) ? gdb : udb;
    int sub = wv & (MSUB - 1);
    int rr = lane & 15, mb = (lane >> 4) * 8;
    float db = dbp[e * RRANK + rr];
#pragma unroll
    for (int v = 0; v < 8; ++v) {
      int row = sub * 16 + mb + v;
      sL[row * LDP + rr]      = (_Float16)(cl[v] + db);
      sL[row * LDP + 16 + rr] = (_Float16)0.0f;
    }
  }
  // Stage rank-16 up-projection weights (K padded to 32 with zeros).
  for (int idx = tid; idx < 32 * 128; idx += 256) {
    int n = idx & 127, k = idx >> 7;
    int wn = n >> 4, ni = n & 15;
    float vg = (k < RRANK) ? guw[(size_t)e * RRANK * IDIM + (size_t)k * IDIM + n0 + n] : 0.0f;
    float vu = (k < RRANK) ? uuw[(size_t)e * RRANK * IDIM + (size_t)k * IDIM + n0 + n] : 0.0f;
    sBg[wn][ni * LDP + k] = (_Float16)vg;
    sBu[wn][ni * LDP + k] = (_Float16)vu;
  }
  __syncthreads();

  v8f lg[MSUB]; v8f lu[MSUB];
  {
    v16h bgf = load_b_frag(sBg[wv], lane);
    v16h buf = load_b_frag(sBu[wv], lane);
#pragma unroll
    for (int s = 0; s < MSUB; ++s) {
      v8f z0 = {}; v8f z1 = {};
      lg[s] = wmma_f16(load_a_frag(sLg + s * 16 * LDP, lane), bgf, z0);
      lu[s] = wmma_f16(load_a_frag(sLu + s * 16 * LDP, lane), buf, z1);
    }
  }

  int col = lane & 15, mb = (lane >> 4) * 8;
  int nglob = n0 + wv * 16 + col;
  float bg   = gbb[e * IDIM + nglob];
  float bu   = ubb[e * IDIM + nglob];
  float ugb  = gub[e * IDIM + nglob];
  float uub_ = uub[e * IDIM + nglob];
  int growbase = row_off[e] + m0;
#pragma unroll
  for (int s = 0; s < MSUB; ++s) {
#pragma unroll
    for (int v = 0; v < 8; ++v) {
      int m = s * 16 + mb + v;
      if (m0 + m < cnt) {
        float gpre = cg[s][v] + bg + SCALING_F * (lg[s][v] + ugb);
        float upre = cu[s][v] + bu + SCALING_F * (lu[s][v] + uub_);
        float h = era_f(gpre) * upre;
        hmid[(size_t)(growbase + m) * IDIM + nglob] = (_Float16)h;
      }
    }
  }
}

// ---------------------------------------------------------------------------
// Kernel 4: Y = h @ down_w + down_b, pre-scaled by the routing weight.
// grid = (HDIM/128, NTOK/BM, NEXP), block = 256.  The f16 A tile (hmid) is
// loaded LDS-direct by the Tensor Data Mover: 2D tile 32x64, data_size=2B,
// LDS padding 4 DWORDs every 16 DWORDs -> exactly the LDP=40-half pitch the
// fragment loaders use; tensor_dim1 = remaining rows so the ragged tail is
// zero-filled by TDM OOB handling.
// ---------------------------------------------------------------------------
__global__ __launch_bounds__(256) void down_kernel(
    const _Float16* __restrict__ hmid,
    const float* __restrict__ dw, const float* __restrict__ db,
    const int* __restrict__ counts, const int* __restrict__ row_off,
    const float* __restrict__ token_w, float* __restrict__ Ybuf) {
  int e   = blockIdx.z;
  int cnt = counts[e];
  int m0  = blockIdx.y * BM;
  if (m0 >= cnt) return;
  int h0   = blockIdx.x * 128;
  int tid  = threadIdx.x;
  int lane = tid & 31;
  int wv   = tid >> 5;

  __shared__ _Float16 sA[BM * LDP];
  __shared__ _Float16 sB[8][16 * LDP];
  __shared__ float sW[BM];

  int grow = row_off[e] + m0;
  for (int i = tid; i < BM; i += 256) {
    int m = m0 + i;
    sW[i] = (m < cnt) ? token_w[e * NTOK + m] : 0.0f;
  }

  // TDM descriptor pieces that do not change across K-steps.
  unsigned lds_off = (unsigned)(uintptr_t)(&sA[0]);   // low 32 bits = LDS addr
  unsigned long long gbase =
      (unsigned long long)(uintptr_t)(hmid + (size_t)grow * IDIM);
  unsigned trows = (unsigned)(cnt - m0);              // rows available (<=2048)

  const float* W = dw + (size_t)e * IDIM * HDIM;
  v8f c[MSUB];
#pragma unroll
  for (int s = 0; s < MSUB; ++s) c[s] = (v8f){};

  __syncthreads();

  for (int k0 = 0; k0 < IDIM; k0 += 32) {
    // Async tensor load of the A tile (f16 hmid rows) into LDS via TDM.
    if (wv == 0) {
      unsigned long long ga = gbase + (unsigned long long)k0 * 2ull;
      v4u g0;
      g0[0] = 1u;                                   // count = 1 valid descriptor
      g0[1] = lds_off;                              // LDS destination address
      g0[2] = (unsigned)(ga & 0xffffffffu);         // global_addr[31:0]
      g0[3] = (unsigned)((ga >> 32) & 0x01ffffffu) | (2u << 30);  // type=2
      v8i g1;
      g1[0] = (int)((1u << 16)    // data_size = 2 bytes
                  | (1u << 20)    // pad_enable
                  | (3u << 22)    // pad_interval code 3 = 16 DWORDs (64 B row)
                  | (3u << 25));  // pad_amount  code 3 = 4 DWORDs (16 B pad)
      g1[1] = (int)(((unsigned)IDIM & 0xffffu) << 16);   // tensor_dim0 lo
      g1[2] = (int)((trows & 0xffffu) << 16);            // dim0 hi=0 | dim1 lo
      g1[3] = (int)(((trows >> 16) & 0xffffu) | (32u << 16)); // dim1 hi | tile_dim0=32
      g1[4] = (int)(64u);                                // tile_dim1=64, tile_dim2=0
      g1[5] = (int)IDIM;                                 // tensor_dim0_stride lo
      g1[6] = 0;                                         // stride hi | dim1_stride lo
      g1[7] = 0;
      v4i gz = {0, 0, 0, 0};
#if defined(__clang_major__) && __clang_major__ >= 23
      v8i gz8 = {0, 0, 0, 0, 0, 0, 0, 0};
      __builtin_amdgcn_tensor_load_to_lds(g0, g1, gz, gz, gz8, 0);
#else
      __builtin_amdgcn_tensor_load_to_lds(g0, g1, gz, gz, 0);
#endif
    }
    // Stage B: 32 K x 128 N down-weights (f32 -> f16), column-major chunks.
    if (k0 + 32 < IDIM) {
      const float* pw = &W[(size_t)(k0 + 32 + (tid >> 3)) * HDIM + h0 + (tid & 7) * 16];
      __builtin_prefetch(pw, 0, 1);
    }
    for (int idx = tid; idx < 32 * 128; idx += 256) {
      int n = idx & 127, k = idx >> 7;
      int wn = n >> 4, ni = n & 15;
      sB[wn][ni * LDP + k] = (_Float16)W[(size_t)(k0 + k) * HDIM + h0 + n];
    }
    if (wv == 0) __builtin_amdgcn_s_wait_tensorcnt(0);
    __syncthreads();

    v16h bf = load_b_frag(sB[wv], lane);
#pragma unroll
    for (int s = 0; s < MSUB; ++s)
      c[s] = wmma_f16(load_a_frag(sA + s * 16 * LDP, lane), bf, c[s]);
    __syncthreads();
  }

  int col = lane & 15, mb = (lane >> 4) * 8;
  int hcol = h0 + wv * 16 + col;
  float bias = db[e * HDIM + hcol];
#pragma unroll
  for (int s = 0; s < MSUB; ++s) {
#pragma unroll
    for (int v = 0; v < 8; ++v) {
      int m = s * 16 + mb + v;
      if (m0 + m < cnt) {
        float y = c[s][v] + bias;
        Ybuf[(size_t)(grow + m) * HDIM + hcol] = sW[m] * y;
      }
    }
  }
}

// ---------------------------------------------------------------------------
// Kernel 5: deterministic 2-way combine per token.
// ---------------------------------------------------------------------------
__global__ __launch_bounds__(256) void combine_kernel(
    const float* __restrict__ Ybuf, const int* __restrict__ row_off,
    const int* __restrict__ pr_e, const int* __restrict__ pr_slot,
    float* __restrict__ out) {
  int idx = blockIdx.x * 256 + threadIdx.x;
  if (idx >= NTOK * HDIM) return;
  int t = idx >> 10;          // HDIM == 1024
  int h = idx & (HDIM - 1);
  int r0 = row_off[pr_e[t * 2 + 0]] + pr_slot[t * 2 + 0];
  int r1 = row_off[pr_e[t * 2 + 1]] + pr_slot[t * 2 + 1];
  out[idx] = Ybuf[(size_t)r0 * HDIM + h] + Ybuf[(size_t)r1 * HDIM + h];
}

// ---------------------------------------------------------------------------
// Host launcher.
// ---------------------------------------------------------------------------
extern "C" void kernel_launch(void* const* d_in, const int* in_sizes, int n_in,
                              void* d_out, int out_size, void* d_ws, size_t ws_size,
                              hipStream_t stream) {
  const float* x    = (const float*)d_in[0];
  const float* rw   = (const float*)d_in[1];
  const float* rb   = (const float*)d_in[2];
  const float* gbw  = (const float*)d_in[3];
  const float* gbb  = (const float*)d_in[4];
  const float* gdw  = (const float*)d_in[5];
  const float* gdb  = (const float*)d_in[6];
  const float* guw  = (const float*)d_in[7];
  const float* gub  = (const float*)d_in[8];
  const float* ubw  = (const float*)d_in[9];
  const float* ubb  = (const float*)d_in[10];
  const float* udw  = (const float*)d_in[11];
  const float* udb  = (const float*)d_in[12];
  const float* uuw  = (const float*)d_in[13];
  const float* uub  = (const float*)d_in[14];
  const float* dw   = (const float*)d_in[15];
  const float* db   = (const float*)d_in[16];
  float* out = (float*)d_out;

  char* ws = (char*)d_ws;
  int*   counts    = (int*)(ws + 0);
  int*   row_off   = (int*)(ws + 64);
  int*   token_idx = (int*)(ws + 128);
  float* token_w   = (float*)(ws + 128 + (size_t)NEXP * NTOK * 4);
  int*   pr_e      = (int*)(ws + 131200);
  int*   pr_slot   = (int*)(ws + 147584);
  float* probs     = (float*)(ws + 163968);
  _Float16* hmid   = (_Float16*)(ws + 262144);
  float* Ybuf      = (float*)(ws + 262144 + (size_t)NTOK * NTOPK * IDIM * 2);

  hipMemsetAsync(ws, 0, 128, stream);  // counts

  router_kernel<<<NTOK / 8, 256, 0, stream>>>(x, rw, rb, counts, token_idx,
                                              token_w, pr_e, pr_slot, probs);
  finalize_kernel<<<1, 32, 0, stream>>>(counts, probs, row_off,
                                        out + (size_t)NTOK * HDIM);
  ffn_hmid_kernel<<<dim3(IDIM / 128, NTOK / BM, NEXP), 256, 0, stream>>>(
      x, gbw, gbb, gdw, gdb, guw, gub, ubw, ubb, udw, udb, uuw, uub,
      counts, row_off, token_idx, hmid);
  down_kernel<<<dim3(HDIM / 128, NTOK / BM, NEXP), 256, 0, stream>>>(
      hmid, dw, db, counts, row_off, token_w, Ybuf);
  combine_kernel<<<(NTOK * HDIM + 255) / 256, 256, 0, stream>>>(
      Ybuf, row_off, pr_e, pr_slot, out);
}
